// ClassifierGNN_33500744909412
// MI455X (gfx1250) — compile-verified, matrix operand order checked by reading, not run on hardware
//
#include <hip/hip_runtime.h>
#include <hip/hip_bf16.h>

// ---------------- model dims ----------------
#define NB     4
#define NPER0  2048
#define NNODE  8192            // NB * NPER0
#define DEG    30
#define NEDGE  (NNODE * DEG)   // 245760
#define NROW3  (NEDGE * 3)     // edge vector rows
#define NSD    100
#define NVD    16

typedef __attribute__((ext_vector_type(16))) __bf16       v16bf;
typedef __attribute__((ext_vector_type(8)))  float        v8f;
typedef __attribute__((ext_vector_type(4)))  unsigned int v4u;

struct U16x16 { v4u lo, hi; };   // 32 bytes == one bf16 WMMA fragment

__device__ __forceinline__ unsigned short f2bf(float f) {
  unsigned int u = __float_as_uint(f);
  u += 0x7fffu + ((u >> 16) & 1u);           // round-to-nearest-even
  return (unsigned short)(u >> 16);
}
__device__ __forceinline__ float bf2f(unsigned short h) {
  return __uint_as_float(((unsigned int)h) << 16);
}

// ---------------- WMMA GEMM: C[M,N] = act(A[M,Kpad](bf16) * Bt[N,Kpad](bf16) + bias) ----------------
// one wave32 computes a 16x32 C tile (two 16x16 WMMA accumulators sharing one A fragment).
// Kpad % 32 == 0. Out-of-range rows/cols are handled by pointer clamping (their results are
// simply never stored), so the inner loop is branch-free: 6x global_load_b128 + 2x WMMA.
__global__ __launch_bounds__(256)
void k_gemm_bf16(const unsigned short* __restrict__ A,
                 const unsigned short* __restrict__ Bt,
                 const float* __restrict__ bias,
                 float* __restrict__ Cf, unsigned short* __restrict__ Cb,
                 int M, int N, int Kpad, int ldc, int nstore, int act)
{
  const int wave = (int)((blockIdx.x * (unsigned)blockDim.x + threadIdx.x) >> 5);
  const int lane = threadIdx.x & 31;
  const int Mt = (M + 15) >> 4;
  const int Np = (nstore + 31) >> 5;
  if (wave >= Mt * Np) return;               // wave-uniform: EXEC stays full for WMMA
  const int mt = wave % Mt;
  const int np = wave / Mt;
  const int hf  = lane >> 4;                 // half-wave selects K sub-range
  const int l15 = lane & 15;
  const int mr  = min(mt * 16 + l15, M - 1);           // clamped: OOB rows never stored
  const int nc0 = min(np * 32 + l15, N - 1);           // clamped: OOB cols never stored
  const int nc1 = min(np * 32 + 16 + l15, N - 1);
  const unsigned short* ar = A  + (size_t)mr  * Kpad + (hf << 3);
  const unsigned short* b0 = Bt + (size_t)nc0 * Kpad + (hf << 3);
  const unsigned short* b1 = Bt + (size_t)nc1 * Kpad + (hf << 3);
  v8f acc0 = {}, acc1 = {};
  for (int kb = 0; kb < Kpad; kb += 32) {
    U16x16 fa, fb0, fb1;
    fa.lo  = *(const v4u*)(ar + kb);  fa.hi  = *(const v4u*)(ar + kb + 16);
    fb0.lo = *(const v4u*)(b0 + kb);  fb0.hi = *(const v4u*)(b0 + kb + 16);
    fb1.lo = *(const v4u*)(b1 + kb);  fb1.hi = *(const v4u*)(b1 + kb + 16);
    v16bf av = __builtin_bit_cast(v16bf, fa);
    acc0 = __builtin_amdgcn_wmma_f32_16x16x32_bf16(
        false, av, false, __builtin_bit_cast(v16bf, fb0), (short)0, acc0, false, false);
    acc1 = __builtin_amdgcn_wmma_f32_16x16x32_bf16(
        false, av, false, __builtin_bit_cast(v16bf, fb1), (short)0, acc1, false, false);
  }
#pragma unroll
  for (int half = 0; half < 2; ++half) {
    const v8f& acc = half ? acc1 : acc0;
    const int tn = np * 32 + half * 16;
#pragma unroll
    for (int r = 0; r < 8; ++r) {
      int m = mt * 16 + r + (hf << 3);
      int n = tn + l15;
      if (m < M && n < nstore) {
        float v = 0.f;
        if (n < N) {
          v = acc[r];
          if (bias) v += bias[n];
          if (act == 1) v = fmaxf(v, 0.f);
        }
        if (Cf) Cf[(size_t)m * ldc + n] = v;
        else    Cb[(size_t)m * ldc + n] = f2bf(v);
      }
    }
  }
}

// ---------------- small helper kernels ----------------
__global__ void k_pack_w(const float* __restrict__ w, unsigned short* __restrict__ o,
                         int N, int K, int Kpad) {
  int idx = blockIdx.x * blockDim.x + threadIdx.x;
  if (idx >= N * Kpad) return;
  int n = idx / Kpad, k = idx % Kpad;
  o[idx] = (k < K) ? f2bf(w[(size_t)n * K + k]) : (unsigned short)0;
}

__global__ void k_cvt_bf16(const float* __restrict__ x, unsigned short* __restrict__ y,
                           int M, int K, int Kpad) {
  int idx = blockIdx.x * blockDim.x + threadIdx.x;
  if (idx >= M * Kpad) return;
  int m = idx / Kpad, k = idx % Kpad;
  y[idx] = (k < K) ? f2bf(x[(size_t)m * K + k]) : (unsigned short)0;
}

// h_V_v [n,nv,3] -> row layout [(i*3+d)*nv + k]
__global__ void k_cvt_hvv(const float* __restrict__ in, float* __restrict__ out, int n, int nv) {
  int idx = blockIdx.x * blockDim.x + threadIdx.x;
  if (idx >= n * nv * 3) return;
  int i = idx / (nv * 3), r = idx % (nv * 3), k = r / 3, d = r % 3;
  out[((size_t)i * 3 + d) * nv + k] = in[idx];
}

// LayerNorm on scalars + RMS-over-channels norm on vectors (optionally with residual)
__global__ __launch_bounds__(128)
void k_gvp_ln(const float* __restrict__ s_in, const float* __restrict__ s_res,
              const float* __restrict__ v_in, const float* __restrict__ v_res,
              float* __restrict__ s_out, float* __restrict__ v_out,
              const float* __restrict__ g, const float* __restrict__ b,
              int rows, int nsd, int nvd)
{
  int row  = (int)((blockIdx.x * (unsigned)blockDim.x + threadIdx.x) >> 5);
  int lane = threadIdx.x & 31;
  if (row >= rows) return;
  const float* si = s_in + (size_t)row * nsd;
  const float* sr = s_res ? s_res + (size_t)row * nsd : nullptr;
  float lsum = 0.f, lsq = 0.f;
  for (int c = lane; c < nsd; c += 32) {
    float x = si[c] + (sr ? sr[c] : 0.f); lsum += x; lsq += x * x;
  }
  for (int o = 16; o; o >>= 1) { lsum += __shfl_xor(lsum, o); lsq += __shfl_xor(lsq, o); }
  float mu = lsum / nsd;
  float rstd = rsqrtf(lsq / nsd - mu * mu + 1e-5f);
  for (int c = lane; c < nsd; c += 32) {
    float x = si[c] + (sr ? sr[c] : 0.f);
    s_out[(size_t)row * nsd + c] = (x - mu) * rstd * g[c] + b[c];
  }
  float accv = 0.f;
  for (int k = lane; k < nvd; k += 32) {
    float sq = 0.f;
    for (int d = 0; d < 3; ++d) {
      size_t idx = ((size_t)row * 3 + d) * nvd + k;
      float x = v_in[idx] + (v_res ? v_res[idx] : 0.f);
      sq += x * x;
    }
    accv += fmaxf(sq, 1e-8f);
  }
  for (int o = 16; o; o >>= 1) accv += __shfl_xor(accv, o);
  float inv = rsqrtf(accv / nvd);
  for (int k = lane; k < nvd; k += 32)
    for (int d = 0; d < 3; ++d) {
      size_t idx = ((size_t)row * 3 + d) * nvd + k;
      float x = v_in[idx] + (v_res ? v_res[idx] : 0.f);
      v_out[idx] = x * inv;
    }
}

// gvp_v: vi=3, h=16 -> cat[n,128] bf16 (s|vn|0), v_out rows [n*3,16]
__global__ void k_gvp_v(const float* __restrict__ s_ln, const float* __restrict__ v_ln,
                        const float* __restrict__ wh, const float* __restrict__ wv,
                        unsigned short* __restrict__ cat, float* __restrict__ v_out, int n)
{
  int i = blockIdx.x * blockDim.x + threadIdx.x;
  if (i >= n) return;
  float vh[3][16];
  for (int d = 0; d < 3; ++d)
    for (int j = 0; j < 16; ++j) {
      float a = 0.f;
      for (int c = 0; c < 3; ++c) a += v_ln[((size_t)i * 3 + d) * 3 + c] * wh[j * 3 + c];
      vh[d][j] = a;
    }
  unsigned short* o = cat + (size_t)i * 128;
  for (int c = 0; c < 100; ++c) o[c] = f2bf(s_ln[(size_t)i * 100 + c]);
  for (int j = 0; j < 16; ++j) {
    float q = vh[0][j]*vh[0][j] + vh[1][j]*vh[1][j] + vh[2][j]*vh[2][j];
    o[100 + j] = f2bf(sqrtf(fmaxf(q, 1e-8f)));
  }
  for (int c = 116; c < 128; ++c) o[c] = 0;
  for (int k = 0; k < 16; ++k)
    for (int d = 0; d < 3; ++d) {
      float a = 0.f;
      for (int j = 0; j < 16; ++j) a += vh[d][j] * wv[k * 16 + j];
      v_out[((size_t)i * 3 + d) * 16 + k] = a;
    }
}

// gvp_e: vi=1,h=1 -> cat[E,64] bf16 (es|vn|0), ev_out rows [E*3]
__global__ void k_gvp_e(const float* __restrict__ es_ln, const float* __restrict__ ev_ln,
                        const float* __restrict__ wh, const float* __restrict__ wv,
                        unsigned short* __restrict__ cat, float* __restrict__ ev_out, int E)
{
  int e = blockIdx.x * blockDim.x + threadIdx.x;
  if (e >= E) return;
  float w0 = wh[0], w1 = wv[0];
  float v0 = ev_ln[(size_t)e*3+0]*w0, v1 = ev_ln[(size_t)e*3+1]*w0, v2 = ev_ln[(size_t)e*3+2]*w0;
  float vn = sqrtf(fmaxf(v0*v0 + v1*v1 + v2*v2, 1e-8f));
  unsigned short* o = cat + (size_t)e * 64;
  for (int c = 0; c < 32; ++c) o[c] = f2bf(es_ln[(size_t)e * 32 + c]);
  o[32] = f2bf(vn);
  for (int c = 33; c < 64; ++c) o[c] = 0;
  ev_out[(size_t)e*3+0] = v0*w1; ev_out[(size_t)e*3+1] = v1*w1; ev_out[(size_t)e*3+2] = v2*w1;
}

// A_mv [3E,64] bf16: row(3e+d) = [v[src](16) | ev | v[dst](16) | 0]
__global__ void k_build_Amv(const float* __restrict__ v, const float* __restrict__ evr,
                            const int* __restrict__ src, const int* __restrict__ dst,
                            unsigned short* __restrict__ A, int E)
{
  int r = blockIdx.x * blockDim.x + threadIdx.x;
  if (r >= 3 * E) return;
  int e = r / 3, d = r % 3;
  int s_ = src[e], t_ = dst[e];
  unsigned short* o = A + (size_t)r * 64;
  for (int k = 0; k < 16; ++k) o[k]      = f2bf(v[((size_t)s_ * 3 + d) * 16 + k]);
  o[16] = f2bf(evr[r]);
  for (int k = 0; k < 16; ++k) o[17 + k] = f2bf(v[((size_t)t_ * 3 + d) * 16 + k]);
  for (int k = 33; k < 64; ++k) o[k] = 0;
}

// A_ms1 [E,288] bf16: [s_src(100)|es(32)|s_dst(100)|vn0(33)|0]
__global__ void k_build_Ams1(const float* __restrict__ s, const unsigned short* __restrict__ esb,
                             const unsigned short* __restrict__ vh,
                             const int* __restrict__ src, const int* __restrict__ dst,
                             unsigned short* __restrict__ A, int E)
{
  int e = blockIdx.x * blockDim.x + threadIdx.x;
  if (e >= E) return;
  int s_ = src[e], t_ = dst[e];
  unsigned short* o = A + (size_t)e * 288;
  for (int c = 0; c < 100; ++c) o[c]       = f2bf(s[(size_t)s_ * 100 + c]);
  for (int c = 0; c < 32;  ++c) o[100 + c] = esb[(size_t)e * 32 + c];
  for (int c = 0; c < 100; ++c) o[132 + c] = f2bf(s[(size_t)t_ * 100 + c]);
  for (int j = 0; j < 33; ++j) {
    float q = 0.f;
    for (int d = 0; d < 3; ++d) { float x = bf2f(vh[((size_t)e * 3 + d) * 64 + j]); q += x * x; }
    o[232 + j] = f2bf(sqrtf(fmaxf(q, 1e-8f)));
  }
  for (int c = 265; c < 288; ++c) o[c] = 0;
}

// generic cat [rows,Kpad] bf16 = [s(ns) | vn(nvh from vh bf16 rows ldvh) | 0]
__global__ void k_build_cat(const float* __restrict__ s, int ns,
                            const unsigned short* __restrict__ vh, int nvh, int ldvh,
                            unsigned short* __restrict__ A, int Kpad, int rows)
{
  int i = blockIdx.x * blockDim.x + threadIdx.x;
  if (i >= rows) return;
  unsigned short* o = A + (size_t)i * Kpad;
  for (int c = 0; c < ns; ++c) o[c] = f2bf(s[(size_t)i * ns + c]);
  for (int j = 0; j < nvh; ++j) {
    float q = 0.f;
    for (int d = 0; d < 3; ++d) { float x = bf2f(vh[((size_t)i * 3 + d) * ldvh + j]); q += x * x; }
    o[ns + j] = f2bf(sqrtf(fmaxf(q, 1e-8f)));
  }
  for (int c = ns + nvh; c < Kpad; ++c) o[c] = 0;
}

// vector gating: y = bf16( x * sigmoid(||x||_d) ), rows grouped by 3 dims
__global__ void k_gate(const float* __restrict__ x, unsigned short* __restrict__ y,
                       int groups, int nv, int ldo)
{
  int i = blockIdx.x * blockDim.x + threadIdx.x;
  if (i >= groups) return;
  for (int k = 0; k < nv; ++k) {
    float a0 = x[((size_t)i*3+0)*nv+k], a1 = x[((size_t)i*3+1)*nv+k], a2 = x[((size_t)i*3+2)*nv+k];
    float nr = sqrtf(fmaxf(a0*a0 + a1*a1 + a2*a2, 1e-8f));
    float sg = 1.f / (1.f + __expf(-nr));
    y[((size_t)i*3+0)*ldo+k] = f2bf(a0*sg);
    y[((size_t)i*3+1)*ldo+k] = f2bf(a1*sg);
    y[((size_t)i*3+2)*ldo+k] = f2bf(a2*sg);
  }
  for (int k = nv; k < ldo; ++k)
    for (int d = 0; d < 3; ++d) y[((size_t)i*3+d)*ldo+k] = 0;
}

// mean over each node's contiguous 30-edge block (scalars)
__global__ void k_seg_mean_s(const float* __restrict__ ms, float* __restrict__ ds, int n)
{
  int idx = blockIdx.x * blockDim.x + threadIdx.x;
  if (idx >= n * 100) return;
  int i = idx / 100, c = idx % 100;
  float s = 0.f;
  for (int t = 0; t < DEG; ++t) s += ms[((size_t)i * DEG + t) * 100 + c];
  ds[idx] = s * (1.f / DEG);
}
// vectors
__global__ void k_seg_mean_v(const float* __restrict__ mv, float* __restrict__ dv, int n)
{
  int idx = blockIdx.x * blockDim.x + threadIdx.x;
  if (idx >= n * 48) return;
  int i = idx / 48, r = idx % 48, d = r / 16, k = r % 16;
  float s = 0.f;
  for (int t = 0; t < DEG; ++t) s += mv[(((size_t)(i * DEG + t)) * 3 + d) * 16 + k];
  dv[((size_t)i * 3 + d) * 16 + k] = s * (1.f / DEG);
}

// gvp_out cat build (vi=16 -> vn(16))
__global__ void k_gvpout(const float* __restrict__ s_ln, const float* __restrict__ v_ln,
                         const float* __restrict__ wh, unsigned short* __restrict__ cat, int n)
{
  int i = blockIdx.x * blockDim.x + threadIdx.x;
  if (i >= n) return;
  unsigned short* o = cat + (size_t)i * 128;
  for (int c = 0; c < 100; ++c) o[c] = f2bf(s_ln[(size_t)i * 100 + c]);
  for (int j = 0; j < 16; ++j) {
    float q = 0.f;
    for (int d = 0; d < 3; ++d) {
      float a = 0.f;
      for (int k = 0; k < 16; ++k) a += v_ln[((size_t)i * 3 + d) * 16 + k] * wh[j * 16 + k];
      q += a * a;
    }
    o[100 + j] = f2bf(sqrtf(fmaxf(q, 1e-8f)));
  }
  for (int c = 116; c < 128; ++c) o[c] = 0;
}

// ---------------- pooling kernels ----------------
__global__ void k_init_edges(const int* __restrict__ ei, int* __restrict__ cs,
                             int* __restrict__ cd, int* __restrict__ em, int E)
{
  int e = blockIdx.x * blockDim.x + threadIdx.x;
  if (e >= E) return;
  cs[e] = ei[e]; cd[e] = ei[E + e]; em[e] = 1;
}
__global__ void k_iota(int* __restrict__ p, int n)
{ int i = blockIdx.x * blockDim.x + threadIdx.x; if (i < n) p[i] = i; }

// wave32 per current node; node i's edges = 30-block of original node cperm[i]
__global__ __launch_bounds__(128)
void k_attn(const float* __restrict__ q, const float* __restrict__ k, const float* __restrict__ v,
            const float* __restrict__ sk, const int* __restrict__ csrc, const int* __restrict__ em,
            const int* __restrict__ cperm, float* __restrict__ out, int ncur)
{
  int i = (int)((blockIdx.x * (unsigned)blockDim.x + threadIdx.x) >> 5);
  int lane = threadIdx.x & 31;
  if (i >= ncur) return;
  int p = cperm[i];
  float mylog = 0.f, m = -1e30f; int anyv = 0;
  for (int t = 0; t < DEG; ++t) {
    int e = p * DEG + t;
    if (em[e]) {
      int s_ = csrc[e];
      float d = 0.f;
      for (int c = lane; c < 100; c += 32) d += q[(size_t)i * 100 + c] * k[(size_t)s_ * 100 + c];
      for (int o = 16; o; o >>= 1) d += __shfl_xor(d, o);
      d *= 0.1f;                               // / sqrt(100)
      if (lane == t) mylog = d;
      m = fmaxf(m, d);
      anyv = 1;
    }
  }
  float denom = 0.f, a0 = 0.f, a1 = 0.f, a2 = 0.f, a3 = 0.f;
  if (anyv) {
    for (int t = 0; t < DEG; ++t) {
      int e = p * DEG + t;
      if (em[e]) {
        float l = __shfl(mylog, t);
        float w = __expf(l - m);
        denom += w;
        int s_ = csrc[e];
        a0 += w * v[(size_t)s_ * 100 + lane];
        a1 += w * v[(size_t)s_ * 100 + lane + 32];
        a2 += w * v[(size_t)s_ * 100 + lane + 64];
        if (lane < 4) a3 += w * v[(size_t)s_ * 100 + lane + 96];
      }
    }
  }
  float inv = 1.f / fmaxf(denom, 1e-16f);
  out[(size_t)i * 100 + lane]      = a0 * inv + sk[(size_t)i * 100 + lane];
  out[(size_t)i * 100 + lane + 32] = a1 * inv + sk[(size_t)i * 100 + lane + 32];
  out[(size_t)i * 100 + lane + 64] = a2 * inv + sk[(size_t)i * 100 + lane + 64];
  if (lane < 4)
    out[(size_t)i * 100 + lane + 96] = a3 * inv + sk[(size_t)i * 100 + lane + 96];
}

__global__ void k_bn(float* __restrict__ x, const float* __restrict__ g, const float* __restrict__ b,
                     const float* __restrict__ m, const float* __restrict__ v, int total)
{
  int idx = blockIdx.x * blockDim.x + threadIdx.x;
  if (idx >= total) return;
  int c = idx % 100;
  x[idx] = (x[idx] - m[c]) * rsqrtf(v[c] + 1e-5f) * g[c] + b[c];
}

__global__ void k_score(const float* __restrict__ x, const float* __restrict__ w,
                        float* __restrict__ s, int n)
{
  int i = blockIdx.x * blockDim.x + threadIdx.x;
  if (i >= n) return;
  float dot = 0.f, wn = 0.f;
  for (int c = 0; c < 100; ++c) { dot += x[(size_t)i * 100 + c] * w[c]; wn += w[c] * w[c]; }
  s[i] = tanhf(dot * rsqrtf(wn));
}

// bitonic top-k per graph (descending score, ascending index tiebreak)
__global__ __launch_bounds__(512)
void k_topk_sort(const float* __restrict__ scores, int* __restrict__ perm, int n_per, int kk)
{
  __shared__ unsigned long long sh[2048];
  int g = blockIdx.x, nt = blockDim.x;
  for (int t = threadIdx.x; t < n_per; t += nt) {
    unsigned u = __float_as_uint(scores[(size_t)g * n_per + t]);
    unsigned mono = (u & 0x80000000u) ? ~u : (u | 0x80000000u);
    sh[t] = ((unsigned long long)(~mono) << 32) | (unsigned)t;
  }
  __syncthreads();
  for (int k2 = 2; k2 <= n_per; k2 <<= 1)
    for (int j = k2 >> 1; j > 0; j >>= 1) {
      for (int t = threadIdx.x; t < n_per; t += nt) {
        int ixj = t ^ j;
        if (ixj > t) {
          bool up = (t & k2) == 0;
          unsigned long long a = sh[t], b = sh[ixj];
          if ((a > b) == up) { sh[t] = b; sh[ixj] = a; }
        }
      }
      __syncthreads();
    }
  for (int t = threadIdx.x; t < kk; t += nt)
    perm[(size_t)g * kk + t] = g * n_per + (int)(sh[t] & 0xffffffffu);
}

__global__ void k_fill_neg1(int* __restrict__ a, int n)
{ int i = blockIdx.x * blockDim.x + threadIdx.x; if (i < n) a[i] = -1; }
__global__ void k_scatter_newidx(const int* __restrict__ perm, int* __restrict__ ni, int nn)
{ int r = blockIdx.x * blockDim.x + threadIdx.x; if (r < nn) ni[perm[r]] = r; }
__global__ void k_gather_x(const float* __restrict__ xo, const float* __restrict__ sc,
                           const int* __restrict__ perm, float* __restrict__ xn, int nn)
{
  int idx = blockIdx.x * blockDim.x + threadIdx.x;
  if (idx >= nn * 100) return;
  int r = idx / 100, c = idx % 100, o = perm[r];
  xn[idx] = xo[(size_t)o * 100 + c] * sc[o];
}
__global__ void k_remap_edges(int* __restrict__ cs, int* __restrict__ cd, int* __restrict__ em,
                              const int* __restrict__ ni, int E)
{
  int e = blockIdx.x * blockDim.x + threadIdx.x;
  if (e >= E) return;
  int ok = em[e];
  int ns = ok ? ni[cs[e]] : -1;
  int nd = ok ? ni[cd[e]] : -1;
  int keep = (ns >= 0 && nd >= 0) ? 1 : 0;
  cs[e] = keep ? ns : 0; cd[e] = keep ? nd : 0; em[e] = keep;
}
__global__ void k_perm_update(const int* __restrict__ po, int* __restrict__ pn,
                              const int* __restrict__ perm, int nn)
{ int r = blockIdx.x * blockDim.x + threadIdx.x; if (r < nn) pn[r] = po[perm[r]]; }

__global__ void k_gfeats(const float* __restrict__ x, float* __restrict__ gbuf, int kk, int poolIdx)
{
  int idx = blockIdx.x * blockDim.x + threadIdx.x;
  if (idx >= NB * 100) return;
  int g = idx / 100, c = idx % 100;
  float mx = -1e30f, sm = 0.f;
  for (int r = 0; r < kk; ++r) {
    float v = x[((size_t)g * kk + r) * 100 + c];
    mx = fmaxf(mx, v); sm += v;
  }
  float* o = gbuf + (size_t)g * 900 + poolIdx * 300;
  o[c] = mx; o[100 + c] = sm / kk; o[200 + c] = sm;
}

// ---------------- host orchestration ----------------
static inline int cdivi(long long a, long long b) { return (int)((a + b - 1) / b); }

static void gemm(hipStream_t st, const unsigned short* A, const unsigned short* Bt,
                 const float* bias, float* Cf, unsigned short* Cb,
                 int M, int N, int Kpad, int ldc, int nstore, int act)
{
  int waves = ((M + 15) / 16) * ((nstore + 31) / 32);
  k_gemm_bf16<<<cdivi(waves, 8), 256, 0, st>>>(A, Bt, bias, Cf, Cb, M, N, Kpad, ldc, nstore, act);
}

extern "C" void kernel_launch(void* const* d_in, const int* in_sizes, int n_in,
                              void* d_out, int out_size, void* d_ws, size_t ws_size,
                              hipStream_t stream)
{
  (void)in_sizes; (void)n_in; (void)out_size; (void)ws_size;
  const int n = NNODE, E = NEDGE, R = NROW3, Rn = n * 3;

  int ic = 0;
  auto F = [&]() -> const float* { return (const float*)d_in[ic++]; };
  const float* hVs = F();
  const float* hVv = F();
  const float* hEs = F();
  const float* hEv = F();
  const int*   ei  = (const int*)d_in[ic++];
  ic++; // graph_indices (implied by layout)

  const float *f0w=F(), *f0b=F(), *f1w=F(), *f1b=F();
  const float *lnv_g=F(), *lnv_b=F();
  const float *gv_wh=F(), *gv_ws=F(), *gv_wsb=F(), *gv_wv=F();
  const float *lne_g=F(), *lne_b=F();
  const float *ge_wh=F(), *ge_ws=F(), *ge_wsb=F(), *ge_wv=F();
  struct Gvp { const float *wh, *ws, *wsb, *wv; };
  struct Layer { Gvp c[3]; const float *ln0g, *ln0b; Gvp ff0, ff1; const float *ln1g, *ln1b; };
  Layer L[3];
  for (int l = 0; l < 3; ++l) {
    for (int j = 0; j < 3; ++j) L[l].c[j] = { F(), F(), F(), F() };
    L[l].ln0g = F(); L[l].ln0b = F();
    L[l].ff0 = { F(), F(), F(), F() };
    L[l].ff1 = { F(), F(), F(), F() };
    L[l].ln1g = F(); L[l].ln1b = F();
  }
  const float *lno_g=F(), *lno_b=F();
  const float *go_wh=F(), *go_ws=F(), *go_wsb=F();
  struct Pool { const float *qw,*qb,*kw,*kb,*vw,*vb,*sw,*sb,*tw,*tb,*bng,*bnb,*bnm,*bnv,*tk; };
  Pool P[3];
  for (int p = 0; p < 3; ++p)
    P[p] = { F(),F(),F(),F(),F(),F(),F(),F(),F(),F(),F(),F(),F(),F(),F() };
  const float *d0w=F(), *d0b=F(), *d1w=F(), *d1b=F(), *d2w=F(), *d2b=F();

  // ---- workspace bump allocator ----
  char* base = (char*)d_ws; size_t off = 0;
  auto alloc = [&](size_t bytes) -> void* {
    void* p = base + off; off = (off + bytes + 255) & ~(size_t)255; return p;
  };
  // big conv arena
  unsigned short* bufMv = (unsigned short*)alloc((size_t)R * 64 * 2);
  unsigned short* bufVh = (unsigned short*)alloc((size_t)R * 64 * 2);
  char*           bufAc = (char*)alloc((size_t)E * 288 * 2);
  unsigned short* bufA  = (unsigned short*)bufAc;
  float* bufS1 = (float*)alloc((size_t)E * 100 * 4);
  float* bufS2 = (float*)alloc((size_t)E * 100 * 4);
  float* bufVo = (float*)alloc((size_t)R * 16 * 4);
  // persistent state
  float* s_cur = (float*)alloc((size_t)n * 100 * 4);
  float* v_cur = (float*)alloc((size_t)Rn * 16 * 4);
  unsigned short* es_bf = (unsigned short*)alloc((size_t)E * 32 * 2);
  float* ev_rows = (float*)alloc((size_t)E * 3 * 4);
  // small node buffers
  float* s_feat = (float*)alloc((size_t)n * 100 * 4);
  float* s_ln   = (float*)alloc((size_t)n * 100 * 4);
  float* hvr    = (float*)alloc((size_t)n * 9 * 4);
  float* v_ln3  = (float*)alloc((size_t)n * 9 * 4);
  unsigned short* cat_v = (unsigned short*)alloc((size_t)n * 128 * 2);
  float* dsb = (float*)alloc((size_t)n * 100 * 4);
  float* dvb = (float*)alloc((size_t)Rn * 16 * 4);
  unsigned short* vbuf = (unsigned short*)alloc((size_t)Rn * 32 * 2);
  unsigned short* vhN  = (unsigned short*)alloc((size_t)Rn * 32 * 2);
  unsigned short* catN = (unsigned short*)alloc((size_t)n * 160 * 2);
  unsigned short* catN2= (unsigned short*)alloc((size_t)n * 448 * 2);
  float* hsb = (float*)alloc((size_t)n * 400 * 4);
  float* voN = (float*)alloc((size_t)Rn * 32 * 4);
  float* s_lno = (float*)alloc((size_t)n * 100 * 4);
  float* v_lno = (float*)alloc((size_t)Rn * 16 * 4);
  unsigned short* cat_o = (unsigned short*)alloc((size_t)n * 128 * 2);
  float* x0 = (float*)alloc((size_t)n * 100 * 4);
  // pooling buffers
  unsigned short* xb = (unsigned short*)alloc((size_t)n * 128 * 2);
  float* qf  = (float*)alloc((size_t)n * 100 * 4);
  float* kf  = (float*)alloc((size_t)n * 100 * 4);
  float* vf  = (float*)alloc((size_t)n * 100 * 4);
  float* skf = (float*)alloc((size_t)n * 100 * 4);
  float* xat = (float*)alloc((size_t)n * 100 * 4);
  float* xtr = (float*)alloc((size_t)n * 100 * 4);
  float* scores = (float*)alloc((size_t)n * 4);
  int* perm   = (int*)alloc((size_t)n * 4);
  int* newidx = (int*)alloc((size_t)n * 4);
  int* cpermA = (int*)alloc((size_t)n * 4);
  int* cpermB = (int*)alloc((size_t)n * 4);
  int* csrc = (int*)alloc((size_t)E * 4);
  int* cdst = (int*)alloc((size_t)E * 4);
  int* emsk = (int*)alloc((size_t)E * 4);
  float* xPa = (float*)alloc((size_t)n * 100 * 4);
  float* xPb = (float*)alloc((size_t)n * 100 * 4);
  float* gbuf = (float*)alloc((size_t)NB * 900 * 4);
  float* h0 = (float*)alloc((size_t)NB * 200 * 4);
  float* h1 = (float*)alloc((size_t)NB * 100 * 4);
  unsigned short* gb16 = (unsigned short*)alloc((size_t)NB * 928 * 2);
  unsigned short* h0b  = (unsigned short*)alloc((size_t)NB * 224 * 2);
  unsigned short* h1b  = (unsigned short*)alloc((size_t)NB * 128 * 2);
  // setup-phase aliases into big arena (consumed before arena first used)
  unsigned short* inA   = bufMv;                                           // [n,32]
  unsigned short* s200b = bufMv + (size_t)n * 32;                          // [n,224]
  float* es_lnf = (float*)bufAc;                                           // [E,32]
  unsigned short* cat_e = (unsigned short*)(bufAc + (size_t)E * 32 * 4);   // [E,64]
  float* ev_lnr = bufS1;                                                   // [E*3]

  // ---- pack weights (bf16, [N, Kpad]) ----
  auto pack = [&](const float* w, int N, int K, int Kpad) -> unsigned short* {
    unsigned short* o = (unsigned short*)alloc((size_t)N * Kpad * 2);
    k_pack_w<<<cdivi((long long)N * Kpad, 256), 256, 0, stream>>>(w, o, N, K, Kpad);
    return o;
  };
  unsigned short* pk_f0 = pack(f0w, 200, 6, 32);
  unsigned short* pk_f1 = pack(f1w, 100, 200, 224);
  unsigned short* pk_gvws = pack(gv_ws, 100, 116, 128);
  unsigned short* pk_gews = pack(ge_ws, 32, 33, 64);
  unsigned short *pk_wh[3][3], *pk_ws[3][3], *pk_wv[3][3];
  unsigned short *pk_fwh0[3], *pk_fws0[3], *pk_fwv0[3];
  unsigned short *pk_fwh1[3], *pk_fws1[3], *pk_fwv1[3];
  for (int l = 0; l < 3; ++l) {
    pk_wh[l][0] = pack(L[l].c[0].wh, 33, 33, 64);
    pk_ws[l][0] = pack(L[l].c[0].ws, 100, 265, 288);
    pk_wv[l][0] = pack(L[l].c[0].wv, 16, 33, 64);
    for (int j = 1; j < 3; ++j) {
      pk_wh[l][j] = pack(L[l].c[j].wh, 16, 16, 32);
      pk_ws[l][j] = pack(L[l].c[j].ws, 100, 116, 128);
      pk_wv[l][j] = pack(L[l].c[j].wv, 16, 16, 32);
    }
    pk_fwh0[l] = pack(L[l].ff0.wh, 32, 16, 32);
    pk_fws0[l] = pack(L[l].ff0.ws, 400, 132, 160);
    pk_fwv0[l] = pack(L[l].ff0.wv, 32, 32, 32);
    pk_fwh1[l] = pack(L[l].ff1.wh, 32, 32, 32);
    pk_fws1[l] = pack(L[l].ff1.ws, 100, 432, 448);
    pk_fwv1[l] = pack(L[l].ff1.wv, 16, 32, 32);
  }
  unsigned short* pk_go = pack(go_ws, 100, 116, 128);
  unsigned short *pk_q[3], *pk_k[3], *pk_v[3], *pk_s[3], *pk_t[3];
  for (int p = 0; p < 3; ++p) {
    pk_q[p] = pack(P[p].qw, 100, 100, 128);
    pk_k[p] = pack(P[p].kw, 100, 100, 128);
    pk_v[p] = pack(P[p].vw, 100, 100, 128);
    pk_s[p] = pack(P[p].sw, 100, 100, 128);
    pk_t[p] = pack(P[p].tw, 100, 100, 128);
  }
  unsigned short* pk_d0 = pack(d0w, 200, 900, 928);
  unsigned short* pk_d1 = pack(d1w, 100, 200, 224);
  unsigned short* pk_d2 = pack(d2w, 10, 100, 128);

  // ---- node featurization ----
  k_cvt_bf16<<<cdivi((long long)n * 32, 256), 256, 0, stream>>>(hVs, inA, n, 6, 32);
  gemm(stream, inA, pk_f0, f0b, nullptr, s200b, n, 200, 32, 224, 224, 1);
  gemm(stream, s200b, pk_f1, f1b, s_feat, nullptr, n, 100, 224, 100, 100, 1);
  k_cvt_hvv<<<cdivi((long long)n * 9, 256), 256, 0, stream>>>(hVv, hvr, n, 3);
  k_gvp_ln<<<cdivi((long long)n * 32, 128), 128, 0, stream>>>(
      s_feat, nullptr, hvr, nullptr, s_ln, v_ln3, lnv_g, lnv_b, n, 100, 3);
  k_gvp_v<<<cdivi(n, 256), 256, 0, stream>>>(s_ln, v_ln3, gv_wh, gv_wv, cat_v, v_cur, n);
  gemm(stream, cat_v, pk_gvws, gv_wsb, s_cur, nullptr, n, 100, 128, 100, 100, 0);

  // ---- edge featurization ----
  k_gvp_ln<<<cdivi((long long)E * 32, 128), 128, 0, stream>>>(
      hEs, nullptr, hEv, nullptr, es_lnf, ev_lnr, lne_g, lne_b, E, 32, 1);
  k_gvp_e<<<cdivi(E, 256), 256, 0, stream>>>(es_lnf, ev_lnr, ge_wh, ge_wv, cat_e, ev_rows, E);
  gemm(stream, cat_e, pk_gews, ge_wsb, nullptr, es_bf, E, 32, 64, 32, 32, 0);

  const int* srcp = ei;
  const int* dstp = ei + E;

  // ---- GVPConv layers ----
  for (int l = 0; l < 3; ++l) {
    k_build_Amv<<<cdivi(R, 256), 256, 0, stream>>>(v_cur, ev_rows, srcp, dstp, bufMv, E);
    gemm(stream, bufMv, pk_wh[l][0], nullptr, nullptr, bufVh, R, 33, 64, 64, 64, 0);
    k_build_Ams1<<<cdivi(E, 256), 256, 0, stream>>>(s_cur, es_bf, bufVh, srcp, dstp, bufA, E);
    gemm(stream, bufA, pk_ws[l][0], L[l].c[0].wsb, bufS1, nullptr, E, 100, 288, 100, 100, 1);
    gemm(stream, bufVh, pk_wv[l][0], nullptr, bufVo, nullptr, R, 16, 64, 16, 16, 0);
    k_gate<<<cdivi(E, 256), 256, 0, stream>>>(bufVo, bufMv, E, 16, 32);
    gemm(stream, bufMv, pk_wh[l][1], nullptr, nullptr, bufVh, R, 16, 32, 32, 32, 0);
    k_build_cat<<<cdivi(E, 256), 256, 0, stream>>>(bufS1, 100, bufVh, 16, 32, bufA, 128, E);
    gemm(stream, bufA, pk_ws[l][1], L[l].c[1].wsb, bufS2, nullptr, E, 100, 128, 100, 100, 1);
    gemm(stream, bufVh, pk_wv[l][1], nullptr, bufVo, nullptr, R, 16, 32, 16, 16, 0);
    k_gate<<<cdivi(E, 256), 256, 0, stream>>>(bufVo, bufMv, E, 16, 32);
    gemm(stream, bufMv, pk_wh[l][2], nullptr, nullptr, bufVh, R, 16, 32, 32, 32, 0);
    k_build_cat<<<cdivi(E, 256), 256, 0, stream>>>(bufS2, 100, bufVh, 16, 32, bufA, 128, E);
    gemm(stream, bufA, pk_ws[l][2], L[l].c[2].wsb, bufS1, nullptr, E, 100, 128, 100, 100, 0);
    gemm(stream, bufVh, pk_wv[l][2], nullptr, bufVo, nullptr, R, 16, 32, 16, 16, 0);
    k_seg_mean_s<<<cdivi((long long)n * 100, 256), 256, 0, stream>>>(bufS1, dsb, n);
    k_seg_mean_v<<<cdivi((long long)n * 48, 256), 256, 0, stream>>>(bufVo, dvb, n);
    k_gvp_ln<<<cdivi((long long)n * 32, 128), 128, 0, stream>>>(
        s_cur, dsb, v_cur, dvb, s_cur, v_cur, L[l].ln0g, L[l].ln0b, n, 100, 16);
    // feed-forward GVPs
    k_cvt_bf16<<<cdivi((long long)Rn * 32, 256), 256, 0, stream>>>(v_cur, vbuf, Rn, 16, 32);
    gemm(stream, vbuf, pk_fwh0[l], nullptr, nullptr, vhN, Rn, 32, 32, 32, 32, 0);
    k_build_cat<<<cdivi(n, 256), 256, 0, stream>>>(s_cur, 100, vhN, 32, 32, catN, 160, n);
    gemm(stream, catN, pk_fws0[l], L[l].ff0.wsb, hsb, nullptr, n, 400, 160, 400, 400, 1);
    gemm(stream, vhN, pk_fwv0[l], nullptr, voN, nullptr, Rn, 32, 32, 32, 32, 0);
    k_gate<<<cdivi(n, 256), 256, 0, stream>>>(voN, vbuf, n, 32, 32);
    gemm(stream, vbuf, pk_fwh1[l], nullptr, nullptr, vhN, Rn, 32, 32, 32, 32, 0);
    k_build_cat<<<cdivi(n, 256), 256, 0, stream>>>(hsb, 400, vhN, 32, 32, catN2, 448, n);
    gemm(stream, catN2, pk_fws1[l], L[l].ff1.wsb, dsb, nullptr, n, 100, 448, 100, 100, 0);
    gemm(stream, vhN, pk_fwv1[l], nullptr, dvb, nullptr, Rn, 16, 32, 16, 16, 0);
    k_gvp_ln<<<cdivi((long long)n * 32, 128), 128, 0, stream>>>(
        s_cur, dsb, v_cur, dvb, s_cur, v_cur, L[l].ln1g, L[l].ln1b, n, 100, 16);
  }

  // ---- W_out ----
  k_gvp_ln<<<cdivi((long long)n * 32, 128), 128, 0, stream>>>(
      s_cur, nullptr, v_cur, nullptr, s_lno, v_lno, lno_g, lno_b, n, 100, 16);
  k_gvpout<<<cdivi(n, 256), 256, 0, stream>>>(s_lno, v_lno, go_wh, cat_o, n);
  gemm(stream, cat_o, pk_go, go_wsb, x0, nullptr, n, 100, 128, 100, 100, 1);

  // ---- TopK pooling blocks ----
  k_init_edges<<<cdivi(E, 256), 256, 0, stream>>>(ei, csrc, cdst, emsk, E);
  k_iota<<<cdivi(n, 256), 256, 0, stream>>>(cpermA, n);
  float* xcur = x0;
  int* pcur = cpermA; int* pnext = cpermB;
  int nper = NPER0;
  for (int p = 0; p < 3; ++p) {
    int ncur = NB * nper;
    int kk = nper / 2;
    k_cvt_bf16<<<cdivi((long long)ncur * 128, 256), 256, 0, stream>>>(xcur, xb, ncur, 100, 128);
    gemm(stream, xb, pk_q[p], P[p].qb, qf, nullptr, ncur, 100, 128, 100, 100, 0);
    gemm(stream, xb, pk_k[p], P[p].kb, kf, nullptr, ncur, 100, 128, 100, 100, 0);
    gemm(stream, xb, pk_v[p], P[p].vb, vf, nullptr, ncur, 100, 128, 100, 100, 0);
    gemm(stream, xb, pk_s[p], P[p].sb, skf, nullptr, ncur, 100, 128, 100, 100, 0);
    k_attn<<<cdivi((long long)ncur * 32, 128), 128, 0, stream>>>(
        qf, kf, vf, skf, csrc, emsk, pcur, xat, ncur);
    k_cvt_bf16<<<cdivi((long long)ncur * 128, 256), 256, 0, stream>>>(xat, xb, ncur, 100, 128);
    gemm(stream, xb, pk_t[p], P[p].tb, xtr, nullptr, ncur, 100, 128, 100, 100, 1);
    k_bn<<<cdivi((long long)ncur * 100, 256), 256, 0, stream>>>(
        xtr, P[p].bng, P[p].bnb, P[p].bnm, P[p].bnv, ncur * 100);
    k_score<<<cdivi(ncur, 256), 256, 0, stream>>>(xtr, P[p].tk, scores, ncur);
    k_topk_sort<<<NB, 512, 0, stream>>>(scores, perm, nper, kk);
    int nnew = NB * kk;
    k_fill_neg1<<<cdivi(ncur, 256), 256, 0, stream>>>(newidx, ncur);
    k_scatter_newidx<<<cdivi(nnew, 256), 256, 0, stream>>>(perm, newidx, nnew);
    float* xnew = (xcur == xPa) ? xPb : xPa;
    k_gather_x<<<cdivi((long long)nnew * 100, 256), 256, 0, stream>>>(xtr, scores, perm, xnew, nnew);
    k_remap_edges<<<cdivi(E, 256), 256, 0, stream>>>(csrc, cdst, emsk, newidx, E);
    k_perm_update<<<cdivi(nnew, 256), 256, 0, stream>>>(pcur, pnext, perm, nnew);
    { int* t = pcur; pcur = pnext; pnext = t; }
    k_gfeats<<<cdivi(NB * 100, 256), 256, 0, stream>>>(xnew, gbuf, kk, p);
    xcur = xnew;
    nper = kk;
  }

  // ---- dense head ----
  k_cvt_bf16<<<cdivi((long long)NB * 928, 256), 256, 0, stream>>>(gbuf, gb16, NB, 900, 928);
  gemm(stream, gb16, pk_d0, d0b, h0, nullptr, NB, 200, 928, 200, 200, 1);
  k_cvt_bf16<<<cdivi((long long)NB * 224, 256), 256, 0, stream>>>(h0, h0b, NB, 200, 224);
  gemm(stream, h0b, pk_d1, d1b, h1, nullptr, NB, 100, 224, 100, 100, 1);
  k_cvt_bf16<<<cdivi((long long)NB * 128, 256), 256, 0, stream>>>(h1, h1b, NB, 100, 128);
  gemm(stream, h1b, pk_d2, d2b, (float*)d_out, nullptr, NB, 10, 128, 10, 10, 0);
}
